// DialogueGCNModel_31172872634709
// MI455X (gfx1250) — compile-verified, matrix operand order checked by reading, not run on hardware
//
#include <hip/hip_runtime.h>
#include <hip/hip_bf16.h>

// ---------------- problem constants (from reference) ----------------
constexpr int B_   = 32;
constexpr int L_   = 256;
constexpr int D_   = 1024;
constexpr int H_   = 128;
constexpr int R_   = 8;
constexpr int NB_  = 30;
constexpr int C_   = 7;
constexpr int MEM_ = D_ + H_;   // 1152
constexpr int N_   = B_ * L_;   // 8192

typedef __attribute__((ext_vector_type(16))) _Float16 v16h;
typedef __attribute__((ext_vector_type(8)))  _Float16 v8h;
typedef __attribute__((ext_vector_type(8)))  float    v8f;

// ---------------- elementwise helpers ----------------
__global__ void zero_f32(float* p, long n) {
    long i = (long)blockIdx.x * blockDim.x + threadIdx.x;
    if (i < n) p[i] = 0.0f;
}

__global__ void cvt_f32_f16(const float* __restrict__ src, _Float16* __restrict__ dst, long n) {
    long i = (long)blockIdx.x * blockDim.x + threadIdx.x;
    if (i < n) dst[i] = (_Float16)src[i];
}

// dst (Nc x Kr, f16) = transpose(src (Kr x Nc, f32))
__global__ void cvt_t_f16(const float* __restrict__ src, _Float16* __restrict__ dst,
                          int Kr, int Nc) {
    long i = (long)blockIdx.x * blockDim.x + threadIdx.x;
    if (i >= (long)Kr * Nc) return;
    int k = (int)(i / Nc);
    int n = (int)(i % Nc);
    dst[(long)n * Kr + k] = (_Float16)src[i];
}

// w_relT[r,h,d] = sum_b comp[r,b] * basis[b,d,h]   (transposed output, f16)
__global__ void wrel_k(const float* __restrict__ comp, const float* __restrict__ basis,
                       _Float16* __restrict__ wrelhT) {
    long i = (long)blockIdx.x * blockDim.x + threadIdx.x;          // over R*D*H
    if (i >= (long)R_ * D_ * H_) return;
    int  r   = (int)(i / ((long)D_ * H_));
    long rem = i % ((long)D_ * H_);
    int  d   = (int)(rem / H_);
    int  h   = (int)(rem % H_);
    float acc = 0.0f;
    #pragma unroll 5
    for (int b = 0; b < NB_; ++b)
        acc += comp[r * NB_ + b] * basis[(long)b * D_ * H_ + rem];
    wrelhT[((long)r * H_ + h) * D_ + d] = (_Float16)acc;
}

// ---------------- WMMA GEMM-NT: C[M,N] = A[M,K] * Bt[N,K]^T (+bias) ----------------
// A row-major (lda), Bt row-major N x K (ldb) -> both operands load as 16B vectors.
// Each wave computes a 16(M) x 64(N) tile: 1 A-fragment feeds 4 WMMAs per K-step.
// HOUT: store _Float16, else float. Batched over blockIdx.z with element strides.
template <bool HOUT>
__global__ void gemm_nt(const _Float16* __restrict__ A, const _Float16* __restrict__ Bt,
                        void* __restrict__ Cv, const float* __restrict__ bias,
                        int M, int Nn, int K, int lda, int ldb, int ldc,
                        size_t sAz, size_t sBz, size_t sCz) {
    const _Float16* Az = A  + (size_t)blockIdx.z * sAz;
    const _Float16* Bz = Bt + (size_t)blockIdx.z * sBz;

    int wave = threadIdx.x >> 5;
    int lane = threadIdx.x & 31;
    int tilesN = Nn >> 6;                             // 64-wide N tiles
    int nTiles = (M >> 4) * tilesN;
    int tile = blockIdx.x * 4 + wave;
    if (tile >= nTiles) return;                       // wave-uniform exit, EXEC stays full
    int tm  = (tile / tilesN) << 4;
    int tn0 = (tile % tilesN) << 6;

    int lh = lane & 15;                               // 0..15 : M (A) / N (B,C)
    int kh = (lane >> 4) << 3;                        // 0 or 8: K-half select (ISA 7.12.2)
    const _Float16* Arow = Az + (size_t)(tm + lh) * lda;
    const _Float16* Brow0 = Bz + (size_t)(tn0 + lh) * ldb;

    v8f acc0 = {}, acc1 = {}, acc2 = {}, acc3 = {};
    for (int kk = 0; kk < K; kk += 32) {
        v16h af;
        {
            v8h a0 = *(const v8h*)(Arow + kk + kh);       // K = kk+kh .. +7
            v8h a1 = *(const v8h*)(Arow + kk + kh + 16);  // K = kk+kh+16 .. +23
            #pragma unroll
            for (int i = 0; i < 8; ++i) { af[i] = a0[i]; af[8 + i] = a1[i]; }
        }
        v8f* accs[4] = { &acc0, &acc1, &acc2, &acc3 };
        #pragma unroll
        for (int j = 0; j < 4; ++j) {
            const _Float16* Brow = Brow0 + (size_t)(j * 16) * ldb;
            v16h bf;
            v8h b0 = *(const v8h*)(Brow + kk + kh);
            v8h b1 = *(const v8h*)(Brow + kk + kh + 16);
            #pragma unroll
            for (int i = 0; i < 8; ++i) { bf[i] = b0[i]; bf[8 + i] = b1[i]; }
            *accs[j] = __builtin_amdgcn_wmma_f32_16x16x32_f16(false, af, false, bf,
                                                              (short)0, *accs[j], false, false);
        }
    }

    int rowoff = (lane >> 4) << 3;                    // lanes 16-31 hold M = v+8
    v8f* accs[4] = { &acc0, &acc1, &acc2, &acc3 };
    #pragma unroll
    for (int j = 0; j < 4; ++j) {
        int col = tn0 + j * 16 + lh;
        float bv = bias ? bias[col] : 0.0f;
        #pragma unroll
        for (int v = 0; v < 8; ++v) {
            int m = tm + rowoff + v;
            float val = (*accs[j])[v] + bv;
            if (HOUT)
                ((_Float16*)Cv)[(size_t)blockIdx.z * sCz + (size_t)m * ldc + col] = (_Float16)val;
            else
                ((float*)Cv)[(size_t)blockIdx.z * sCz + (size_t)m * ldc + col] = val;
        }
    }
}

// ---------------- graph scatter ----------------
__global__ void deg_k(const int* __restrict__ edst, float* __restrict__ deg, int E) {
    int e = blockIdx.x * blockDim.x + threadIdx.x;
    if (e < E) atomicAdd(&deg[edst[e]], 1.0f);
}

// dst[edst[e], :] += srcdata[(etype? etype[e]*relStride:0) + esrc[e]*H + h]
__global__ void scatter_vec(const float* __restrict__ srcdata,
                            const int* __restrict__ esrc, const int* __restrict__ edst,
                            const int* __restrict__ etype, size_t relStride,
                            float* __restrict__ dst, int E) {
    int e = blockIdx.x;
    int h = threadIdx.x;
    if (e >= E) return;
    size_t base = etype ? (size_t)etype[e] * relStride : 0;
    atomicAdd(&dst[(size_t)edst[e] * H_ + h], srcdata[base + (size_t)esrc[e] * H_ + h]);
}

// out1 = agg/deg + roottmp + b1  (f32 + f16 copies)
__global__ void rgcn_combine(const float* __restrict__ agg, const float* __restrict__ deg,
                             const float* __restrict__ roottmp, const float* __restrict__ b1,
                             float* __restrict__ out1, _Float16* __restrict__ out1h) {
    long i = (long)blockIdx.x * blockDim.x + threadIdx.x;
    if (i >= (long)N_ * H_) return;
    int node = (int)(i / H_);
    int h    = (int)(i % H_);
    float v = agg[i] / deg[node] + roottmp[i] + b1[h];
    out1[i]  = v;
    out1h[i] = (_Float16)v;
}

// emotions (node-major) -> emoh[q=t*B+b][m]  and  emoT[b][m][t]  (both f16)
__global__ void concat_k(const float* __restrict__ x, const float* __restrict__ tmpA,
                         const float* __restrict__ tmpB, const float* __restrict__ b_rel2,
                         _Float16* __restrict__ emoh, _Float16* __restrict__ emoT) {
    long i = (long)blockIdx.x * blockDim.x + threadIdx.x;
    if (i >= (long)N_ * MEM_) return;
    int node = (int)(i / MEM_);
    int m    = (int)(i % MEM_);
    int b = node / L_, t = node % L_;
    float v;
    if (m < D_) v = x[(size_t)node * D_ + m];
    else {
        int h = m - D_;
        v = tmpA[(size_t)node * H_ + h] + b_rel2[h] + tmpB[(size_t)node * H_ + h];
    }
    _Float16 hv = (_Float16)v;
    emoh[(size_t)(t * B_ + b) * MEM_ + m] = hv;
    emoT[((size_t)b * MEM_ + m) * L_ + t] = hv;
}

// per (t,b) row: alpha = renorm( softmax(tanh(raw*u^2)) * u )  -> f16
__global__ void attn_softmax(const float* __restrict__ scores, const float* __restrict__ umask,
                             _Float16* __restrict__ alphah) {
    int p = blockIdx.x;            // p = t*B + b
    int b = p % B_;
    int s = threadIdx.x;           // 0..255
    __shared__ float red[L_];
    float u   = umask[b * L_ + s];
    float raw = scores[(size_t)p * L_ + s];
    float v   = tanhf(raw * u * u);

    red[s] = v; __syncthreads();
    for (int o = L_ / 2; o > 0; o >>= 1) { if (s < o) red[s] = fmaxf(red[s], red[s + o]); __syncthreads(); }
    float mx = red[0]; __syncthreads();

    float e = __expf(v - mx);
    red[s] = e; __syncthreads();
    for (int o = L_ / 2; o > 0; o >>= 1) { if (s < o) red[s] += red[s + o]; __syncthreads(); }
    float a = e / red[0] * u; __syncthreads();

    red[s] = a; __syncthreads();
    for (int o = L_ / 2; o > 0; o >>= 1) { if (s < o) red[s] += red[s + o]; __syncthreads(); }
    a = a / (red[0] + 1e-20f);
    alphah[(size_t)p * L_ + s] = (_Float16)a;
}

// logits + log_softmax; row p = t*B+b of hid, output row b*L+t
__global__ void final_k(const float* __restrict__ hid, const float* __restrict__ b_lin,
                        const float* __restrict__ w_fc, const float* __restrict__ b_fc,
                        float* __restrict__ out) {
    int p = blockIdx.x * blockDim.x + threadIdx.x;
    if (p >= N_) return;
    int t = p / B_, b = p % B_;
    float lg[C_];
    #pragma unroll
    for (int c = 0; c < C_; ++c) lg[c] = b_fc[c];
    for (int h = 0; h < H_; ++h) {
        float hv = hid[(size_t)p * H_ + h] + b_lin[h];
        hv = fmaxf(hv, 0.0f);
        #pragma unroll
        for (int c = 0; c < C_; ++c) lg[c] += hv * w_fc[h * C_ + c];
    }
    float mx = lg[0];
    #pragma unroll
    for (int c = 1; c < C_; ++c) mx = fmaxf(mx, lg[c]);
    float sum = 0.0f;
    #pragma unroll
    for (int c = 0; c < C_; ++c) sum += __expf(lg[c] - mx);
    float lse = __logf(sum);
    size_t orow = (size_t)(b * L_ + t) * C_;
    #pragma unroll
    for (int c = 0; c < C_; ++c) out[orow + c] = lg[c] - mx - lse;
}

// ---------------- host-side orchestration ----------------
extern "C" void kernel_launch(void* const* d_in, const int* in_sizes, int n_in,
                              void* d_out, int out_size, void* d_ws, size_t ws_size,
                              hipStream_t stream) {
    const float* x       = (const float*)d_in[0];
    const int*   esrc    = (const int*)  d_in[1];
    const int*   edst    = (const int*)  d_in[2];
    const int*   etype   = (const int*)  d_in[3];
    const float* umask   = (const float*)d_in[4];
    const float* basis   = (const float*)d_in[5];
    const float* comp    = (const float*)d_in[6];
    const float* w_root1 = (const float*)d_in[7];
    const float* b1      = (const float*)d_in[8];
    const float* w_rel2  = (const float*)d_in[9];
    const float* b_rel2  = (const float*)d_in[10];
    const float* w_root2 = (const float*)d_in[11];
    const float* w_t     = (const float*)d_in[12];
    const float* b_t     = (const float*)d_in[13];
    const float* w_lin   = (const float*)d_in[14];
    const float* b_lin   = (const float*)d_in[15];
    const float* w_fc    = (const float*)d_in[16];
    const float* b_fc    = (const float*)d_in[17];
    const int E = in_sizes[1];

    // ---- workspace carve-up (aliased arenas) ----
    char* w = (char*)d_ws;
    size_t off = 0;
    auto take = [&](size_t bytes) -> char* {
        char* p = w + off;
        off += (bytes + 255) & ~(size_t)255;
        return p;
    };
    // arena1: xh (N*D f16) aliased with Xch (N*MEM f16)  [xh dead before Xch written]
    _Float16* xh   = (_Float16*)take((size_t)N_ * MEM_ * 2);
    _Float16* Xch  = xh;
    // arena2: xr (R*N*H f32) aliased with emoh (N*MEM f16)  [xr dead before concat]
    float*    xr   = (float*)take((size_t)R_ * N_ * H_ * 4);
    _Float16* emoh = (_Float16*)xr;
    _Float16* wrelhT   = (_Float16*)take((size_t)R_ * D_ * H_ * 2);   // (R,H,D)
    _Float16* wroot1hT = (_Float16*)take((size_t)D_ * H_ * 2);        // (H,D)
    _Float16* wrel2hT  = (_Float16*)take((size_t)H_ * H_ * 2);        // (H,H)
    _Float16* wroot2hT = (_Float16*)take((size_t)H_ * H_ * 2);        // (H,H)
    _Float16* wthT     = (_Float16*)take((size_t)MEM_ * MEM_ * 2);    // (MEM,MEM)
    _Float16* wlinhT   = (_Float16*)take((size_t)MEM_ * H_ * 2);      // (H,MEM)
    // contiguous atomic-target region: agg | deg | nb  (zeroed every call)
    float* agg = (float*)take(((size_t)N_ * H_ + N_ + (size_t)N_ * H_) * 4);
    float* deg = agg + (size_t)N_ * H_;
    float* nb  = deg + N_;
    float*    roottmp = (float*)take((size_t)N_ * H_ * 4);
    float*    out1    = (float*)take((size_t)N_ * H_ * 4);
    _Float16* out1h   = (_Float16*)take((size_t)N_ * H_ * 2);
    _Float16* nbh     = (_Float16*)take((size_t)N_ * H_ * 2);
    float*    tmpA    = (float*)take((size_t)N_ * H_ * 4);
    float*    tmpB    = (float*)take((size_t)N_ * H_ * 4);
    float*    scores  = (float*)take((size_t)N_ * L_ * 4);
    _Float16* alphah  = (_Float16*)take((size_t)N_ * L_ * 2);
    _Float16* atth    = (_Float16*)take((size_t)N_ * MEM_ * 2);
    _Float16* emoT    = (_Float16*)take((size_t)N_ * MEM_ * 2);       // (B,MEM,L)
    float*    hid     = (float*)take((size_t)N_ * H_ * 4);

    auto blk = [](long n, int t) { return (int)((n + t - 1) / t); };
    auto gemm = [&](bool hout, const _Float16* A, const _Float16* Bt, void* C,
                    const float* bias, int M, int Nn, int K, int lda, int ldb, int ldc,
                    size_t sA, size_t sB, size_t sC, int Z) {
        int tiles = (M / 16) * (Nn / 64);
        dim3 g((tiles + 3) / 4, 1, Z);
        if (hout) gemm_nt<true ><<<g, 128, 0, stream>>>(A, Bt, C, bias, M, Nn, K, lda, ldb, ldc, sA, sB, sC);
        else      gemm_nt<false><<<g, 128, 0, stream>>>(A, Bt, C, bias, M, Nn, K, lda, ldb, ldc, sA, sB, sC);
    };

    // 0) zero atomic targets (agg|deg|nb are contiguous)
    long nzero = (long)N_ * H_ * 2 + N_;
    zero_f32<<<blk(nzero, 256), 256, 0, stream>>>(agg, nzero);

    // 1) f32 -> f16 conversions (weights transposed so every GEMM is NT / vector loads)
    cvt_f32_f16<<<blk((long)N_ * D_, 256), 256, 0, stream>>>(x, xh, (long)N_ * D_);
    cvt_t_f16<<<blk((long)D_ * H_, 256), 256, 0, stream>>>(w_root1, wroot1hT, D_, H_);
    cvt_t_f16<<<blk((long)H_ * H_, 256), 256, 0, stream>>>(w_rel2,  wrel2hT,  H_, H_);
    cvt_t_f16<<<blk((long)H_ * H_, 256), 256, 0, stream>>>(w_root2, wroot2hT, H_, H_);
    cvt_t_f16<<<blk((long)MEM_ * MEM_, 256), 256, 0, stream>>>(w_t, wthT, MEM_, MEM_);
    cvt_t_f16<<<blk((long)MEM_ * H_, 256), 256, 0, stream>>>(w_lin, wlinhT, MEM_, H_);

    // 2) basis decomposition: w_rel^T (R,H,D)
    wrel_k<<<blk((long)R_ * D_ * H_, 256), 256, 0, stream>>>(comp, basis, wrelhT);

    // 3) xr[r] = x @ w_rel[r]   (batched over r)   M=8192 N=128 K=1024
    gemm(false, xh, wrelhT, xr, nullptr, N_, H_, D_, D_, D_, H_,
         0, (size_t)H_ * D_, (size_t)N_ * H_, R_);

    // 4) roottmp = x @ w_root1
    gemm(false, xh, wroot1hT, roottmp, nullptr, N_, H_, D_, D_, D_, H_, 0, 0, 0, 1);

    // 5) degree + relational message aggregation
    deg_k<<<blk(E, 256), 256, 0, stream>>>(edst, deg, E);
    scatter_vec<<<dim3(E), H_, 0, stream>>>(xr, esrc, edst, etype, (size_t)N_ * H_, agg, E);

    // 6) out1 = agg/deg + roottmp + b1
    rgcn_combine<<<blk((long)N_ * H_, 256), 256, 0, stream>>>(agg, deg, roottmp, b1, out1, out1h);

    // 7) GraphConv: nb = segment_sum(out1[src]); out2 = nb@w_rel2 + b_rel2 + out1@w_root2
    scatter_vec<<<dim3(E), H_, 0, stream>>>(out1, esrc, edst, nullptr, 0, nb, E);
    cvt_f32_f16<<<blk((long)N_ * H_, 256), 256, 0, stream>>>(nb, nbh, (long)N_ * H_);
    gemm(false, nbh,   wrel2hT,  tmpA, nullptr, N_, H_, H_, H_, H_, H_, 0, 0, 0, 1);
    gemm(false, out1h, wroot2hT, tmpB, nullptr, N_, H_, H_, H_, H_, H_, 0, 0, 0, 1);

    // 8) emotions -> emoh (t*B+b major) and emoT (b,m,s)   (emoh overwrites xr arena)
    concat_k<<<blk((long)N_ * MEM_, 256), 256, 0, stream>>>(x, tmpA, tmpB, b_rel2, emoh, emoT);

    // 9) Xc = M @ w_t + b_t  -> f16  (overwrites xh arena)   M=8192 N=1152 K=1152
    gemm(true, emoh, wthT, Xch, b_t, N_, MEM_, MEM_, MEM_, MEM_, MEM_, 0, 0, 0, 1);

    // 10) scores[t,b,s] = Xc[t,b,:] . M[s,b,:]   (batched over b)
    gemm(false, Xch, emoh, scores, nullptr, L_, L_, MEM_,
         B_ * MEM_, B_ * MEM_, B_ * L_, (size_t)MEM_, (size_t)MEM_, (size_t)L_, B_);

    // 11) masked tanh-softmax -> alpha (f16)
    attn_softmax<<<dim3(N_), L_, 0, stream>>>(scores, umask, alphah);

    // 12) att = alpha @ M  -> f16  via emoT   M=256 N=1152 K=256 (batched over b)
    gemm(true, alphah, emoT, atth, nullptr, L_, MEM_, L_,
         B_ * L_, L_, B_ * MEM_, (size_t)L_, (size_t)MEM_ * L_, (size_t)MEM_, B_);

    // 13) hid = att @ w_lin   (bias+relu fused into final)   M=8192 N=128 K=1152
    gemm(false, atth, wlinhT, hid, nullptr, N_, H_, MEM_, MEM_, MEM_, H_, 0, 0, 0, 1);

    // 14) logits + log_softmax -> d_out (N, C) in b-major node order
    final_k<<<blk(N_, 256), 256, 0, stream>>>(hid, b_lin, w_fc, b_fc, (float*)d_out);
}
